// RecursiveGated_57234734187258
// MI455X (gfx1250) — matrix operand order repaired and checked by reading, compiled, not run on hardware
//
#include <hip/hip_runtime.h>
#include <hip/hip_bf16.h>
#include <math.h>

// ---------------------------------------------------------------------------
// Problem constants (from reference): B=256, D=9, F=8, H=256, N=1023, L=512
// ---------------------------------------------------------------------------
#define BB   256
#define NN   1023
#define LL   512
#define HH   256
#define FF   8
#define K3H  768   // 3*H
#define K4H  1024  // 4*H

typedef __attribute__((ext_vector_type(16))) __bf16 v16bf;
typedef __attribute__((ext_vector_type(8)))  __bf16 v8bf;
typedef __attribute__((ext_vector_type(8)))  float  v8f;

// ---------------------------------------------------------------------------
// WMMA fragment helpers (wave32, 16x16x32 bf16, f32 accum)
//
// A (16x32 bf16) per-lane layout (this code's contiguous k order, not the ISA table order):
//   lane L: row M = L&15 ; K = kbase + (L>>4)*16 .. +15  (16 contiguous)
// B (32x16 bf16) per-lane layout:
//   lane L: col N = L&15 ; K = kbase + (L>>4)*16 .. +15  (16 contiguous)
// Since out_j = sum_k A_k * W[j,k], column j of B == row j of W (contiguous).
// C/D (16x16 f32): vgpr v, lane L: M = v + 8*(L>>4), N = L&15.
// ---------------------------------------------------------------------------
__device__ __forceinline__ v16bf lds_a_frag(const __bf16* tile, int stride,
                                            int kbase, int lane) {
  int m  = lane & 15;
  int hi = lane >> 4;
  const __bf16* p = tile + m * stride + kbase + hi * 16;
  union { v16bf v; v8bf h[2]; } u;
  u.h[0] = *(const v8bf*)(p);
  u.h[1] = *(const v8bf*)(p + 8);
  return u.v;
}

__device__ __forceinline__ v16bf glb_b_frag(const __bf16* W, int ldw,
                                            int ntile, int kbase, int lane) {
  int n  = lane & 15;
  int kk = kbase + (lane >> 4) * 16;
  return *(const v16bf*)(W + (size_t)(ntile * 16 + n) * ldw + kk);
}

// Two M-tiles (rows 0..15 and 16..31 of the LDS tile) share each B fragment:
// 2 WMMAs per (1 global B + 2 LDS A) fetch.
struct AccPair { v8f m0, m1; };

template <int KTOT>
__device__ __forceinline__ AccPair wmma_chain2(const __bf16* atile, int astride,
                                               const __bf16* W, int ldw,
                                               int ntile, int lane) {
  v8f acc0 = {}, acc1 = {};
#pragma unroll 4
  for (int k = 0; k < KTOT; k += 32) {
    v16bf b  = glb_b_frag(W, ldw, ntile, k, lane);
    v16bf a0 = lds_a_frag(atile, astride, k, lane);
    v16bf a1 = lds_a_frag(atile + 16 * astride, astride, k, lane);
    acc0 = __builtin_amdgcn_wmma_f32_16x16x32_bf16(
        false, a0, false, b, (short)0, acc0, false, false);
    acc1 = __builtin_amdgcn_wmma_f32_16x16x32_bf16(
        false, a1, false, b, (short)0, acc1, false, false);
  }
  AccPair r;
  r.m0 = acc0;
  r.m1 = acc1;
  return r;
}

__device__ __forceinline__ float fast_sigmoid(float x) {
  return __builtin_amdgcn_rcpf(1.0f + __expf(-x));  // v_rcp_f32 + v_exp_f32
}

// ---------------------------------------------------------------------------
// f32 -> bf16 weight conversion
// ---------------------------------------------------------------------------
__global__ void cvt_bf16_kernel(const float* __restrict__ src,
                                __bf16* __restrict__ dst, int n) {
  int i = blockIdx.x * blockDim.x + threadIdx.x;
  if (i < n) dst[i] = (__bf16)src[i];
}

// ---------------------------------------------------------------------------
// Leaf projection: h_leaf[b][i][h] = data[b][511+i][:8] . Wu[h][:8] + bu[h]
// grid = B*512 blocks (one node each), 256 threads (one output col each)
// ---------------------------------------------------------------------------
__global__ __launch_bounds__(HH) void leaf_kernel(
    const float* __restrict__ data, const float* __restrict__ Wu,
    const float* __restrict__ bu, __bf16* __restrict__ h_leaf) {
  int blk = blockIdx.x;
  int b = blk >> 9;  // / 512
  int i = blk & 511;
  __shared__ float d[FF];
  if (threadIdx.x < FF)
    d[threadIdx.x] = data[((size_t)b * NN + (LL - 1 + i)) * FF + threadIdx.x];
  __syncthreads();
  int h = threadIdx.x;
  const float* w = Wu + h * FF;
  float s = bu[h];
#pragma unroll
  for (int f = 0; f < FF; ++f) s = fmaf(d[f], w[f], s);
  h_leaf[((size_t)b * LL + i) * HH + h] = (__bf16)s;
}

// ---------------------------------------------------------------------------
// Fused per-level kernel. One block = 32 rows of (batch,node), 512 threads.
//   hhu  = [parent_u | h_l | h_r]            (cols 256..1023 of hx)
//   r    = sigmoid(hhu . Wr^T + br)          (GEMM1, K=768, Nout=768)
//   rh   = r * hhu                           (-> union.rh, bf16)
//   hnew = relu(rh . Wh^T + bh)              (GEMM2, K=768, Nout=256) -> hx[:,0:256]
//   hhhu = hx (cols 0..1023)
//   zlin = hhhu . Wz^T + bz                  (GEMM3, K=1024, Nout=1024) -> union.zb
//   h    = sum_g softmax_g(zlin) * hhhu_g    -> h_next (bf16)
// LDS: hx 66KB + union(rh 49KB / zb 131KB) ~= 197KB (<= 320KB/WG on CDNA5).
// ---------------------------------------------------------------------------
#define MT   32          // rows per block (two 16-row WMMA M-tiles)
#define HXW  (1024 + 8)
#define RHW  (768 + 8)
#define ZBW  (1024 + 4)

__global__ __launch_bounds__(512) void level_kernel(
    const float* __restrict__ data, const float* __restrict__ Wu,
    const float* __restrict__ bu,
    const __bf16* __restrict__ Wr, const float* __restrict__ br,
    const __bf16* __restrict__ Wh, const float* __restrict__ bh,
    const __bf16* __restrict__ Wz, const float* __restrict__ bz,
    const __bf16* __restrict__ h_prev, __bf16* __restrict__ h_next,
    int n, int log2n) {
  __shared__ __align__(16) __bf16 hx[MT][HXW];
  union Sh { __bf16 rh[MT][RHW]; float zb[MT][ZBW]; };
  __shared__ __align__(16) Sh us;

  const int t    = threadIdx.x;
  const int lane = t & 31;
  const int wave = t >> 5;   // 0..15
  const int row0 = blockIdx.x * MT;

  // ---- Phase 0a: parent u tile -> hx[:, 256..511] ------------------------
  {
    int m  = t >> 4;         // 0..31 : row within tile
    int c0 = (t & 15) * 16;  // 16 output cols per thread
    int row = row0 + m;
    int b = row >> log2n;
    int i = row & (n - 1);
    const float* drow = data + ((size_t)b * NN + (n - 1 + i)) * FF;
    float d[FF];
#pragma unroll
    for (int f = 0; f < FF; ++f) d[f] = drow[f];
#pragma unroll 4
    for (int c = 0; c < 16; ++c) {
      int h = c0 + c;
      const float* w = Wu + h * FF;
      float s = bu[h];
#pragma unroll
      for (int f = 0; f < FF; ++f) s = fmaf(d[f], w[f], s);
      hx[m][256 + h] = (__bf16)s;
    }
  }
  // ---- Phase 0b: children (h_l | h_r) -> hx[:, 512..1023] ----------------
  // children 2i, 2i+1 are contiguous in h_prev: 512 bf16 straight copy.
  {
    int m  = t >> 4;
    int c0 = (t & 15) * 32;
    int row = row0 + m;
    int b = row >> log2n;
    int i = row & (n - 1);
    const __bf16* src = h_prev + ((size_t)b * (2 * n) + 2 * i) * HH;
#pragma unroll
    for (int c = 0; c < 32; c += 8)
      *(v8bf*)&hx[m][512 + c0 + c] = *(const v8bf*)(src + c0 + c);
  }
  __syncthreads();

  // ---- GEMM1: r = sigmoid(hhu.Wr^T + br); rh = r*hhu ---------------------
  for (int nt = wave; nt < K3H / 16; nt += 16) {
    AccPair acc = wmma_chain2<K3H>(&hx[0][256], HXW, Wr, K3H, nt, lane);
    int ncol  = nt * 16 + (lane & 15);
    int mbase = (lane >> 4) * 8;
    float bias = br[ncol];
#pragma unroll
    for (int v = 0; v < 8; ++v) {
      int m0 = mbase + v, m1 = m0 + 16;
      float r0 = fast_sigmoid(acc.m0[v] + bias);
      float r1 = fast_sigmoid(acc.m1[v] + bias);
      us.rh[m0][ncol] = (__bf16)(r0 * (float)hx[m0][256 + ncol]);
      us.rh[m1][ncol] = (__bf16)(r1 * (float)hx[m1][256 + ncol]);
    }
  }
  __syncthreads();

  // ---- GEMM2: hnew = relu(rh.Wh^T + bh) -> hx[:, 0..255] -----------------
  for (int nt = wave; nt < HH / 16; nt += 16) {
    AccPair acc = wmma_chain2<K3H>(&us.rh[0][0], RHW, Wh, K3H, nt, lane);
    int ncol  = nt * 16 + (lane & 15);
    int mbase = (lane >> 4) * 8;
    float bias = bh[ncol];
#pragma unroll
    for (int v = 0; v < 8; ++v) {
      hx[mbase + v][ncol]      = (__bf16)fmaxf(acc.m0[v] + bias, 0.0f);
      hx[mbase + v + 16][ncol] = (__bf16)fmaxf(acc.m1[v] + bias, 0.0f);
    }
  }
  __syncthreads();

  // ---- GEMM3: zlin = hhhu.Wz^T + bz -> us.zb (f32) -----------------------
  // NOTE: zb overlaps rh storage; rh is dead after GEMM2 (barrier above).
  for (int nt = wave; nt < K4H / 16; nt += 16) {
    AccPair acc = wmma_chain2<K4H>(&hx[0][0], HXW, Wz, K4H, nt, lane);
    int ncol  = nt * 16 + (lane & 15);
    int mbase = (lane >> 4) * 8;
    float bias = bz[ncol];
#pragma unroll
    for (int v = 0; v < 8; ++v) {
      us.zb[mbase + v][ncol]      = acc.m0[v] + bias;
      us.zb[mbase + v + 16][ncol] = acc.m1[v] + bias;
    }
  }
  __syncthreads();

  // ---- Softmax over 4 groups + mix -> h_next -----------------------------
  {
    int m  = t >> 4;
    int c0 = (t & 15) * 16;
    int row = row0 + m;
    int b = row >> log2n;
    int i = row & (n - 1);
    __bf16* dst = h_next + ((size_t)b * n + i) * HH;
#pragma unroll 4
    for (int c = c0; c < c0 + 16; ++c) {
      float z0 = us.zb[m][c],       z1 = us.zb[m][256 + c];
      float z2 = us.zb[m][512 + c], z3 = us.zb[m][768 + c];
      float mx = fmaxf(fmaxf(z0, z1), fmaxf(z2, z3));
      float e0 = __expf(z0 - mx), e1 = __expf(z1 - mx);
      float e2 = __expf(z2 - mx), e3 = __expf(z3 - mx);
      float inv = __builtin_amdgcn_rcpf(e0 + e1 + e2 + e3);
      float hv = (e0 * (float)hx[m][c]       + e1 * (float)hx[m][256 + c] +
                  e2 * (float)hx[m][512 + c] + e3 * (float)hx[m][768 + c]) * inv;
      dst[c] = (__bf16)hv;
    }
  }
}

// ---------------------------------------------------------------------------
// Head: out[b] = h0[b,0,:] . Wp[0,:] + bp   (one wave32 per batch element)
// ---------------------------------------------------------------------------
__global__ __launch_bounds__(32) void head_kernel(
    const __bf16* __restrict__ h0, const float* __restrict__ Wp,
    const float* __restrict__ bp, float* __restrict__ out) {
  int b = blockIdx.x;
  int lane = threadIdx.x;
  float s = 0.0f;
  for (int c = lane; c < HH; c += 32)
    s = fmaf((float)h0[(size_t)b * HH + c], Wp[c], s);
#pragma unroll
  for (int o = 16; o > 0; o >>= 1) s += __shfl_xor(s, o, 32);
  if (lane == 0) out[b] = s + bp[0];
}

// ---------------------------------------------------------------------------
// Launch
// inputs: data, Wu, bu, Wr, br, Wh, bh, Wz, bz, Wp, bp
// ---------------------------------------------------------------------------
extern "C" void kernel_launch(void* const* d_in, const int* in_sizes, int n_in,
                              void* d_out, int out_size, void* d_ws,
                              size_t ws_size, hipStream_t stream) {
  const float* data = (const float*)d_in[0];
  const float* Wu   = (const float*)d_in[1];
  const float* bu   = (const float*)d_in[2];
  const float* Wr   = (const float*)d_in[3];
  const float* br   = (const float*)d_in[4];
  const float* Wh   = (const float*)d_in[5];
  const float* bh   = (const float*)d_in[6];
  const float* Wz   = (const float*)d_in[7];
  const float* bz   = (const float*)d_in[8];
  const float* Wp   = (const float*)d_in[9];
  const float* bp   = (const float*)d_in[10];

  // workspace layout (256B aligned)
  char* ws = (char*)d_ws;
  size_t off = 0;
  auto carve = [&](size_t bytes) {
    size_t o = off;
    off = (off + bytes + 255) & ~(size_t)255;
    return o;
  };
  __bf16* Wr_bf = (__bf16*)(ws + carve((size_t)K3H * K3H * 2));
  __bf16* Wh_bf = (__bf16*)(ws + carve((size_t)HH * K3H * 2));
  __bf16* Wz_bf = (__bf16*)(ws + carve((size_t)K4H * K4H * 2));
  __bf16* bufA  = (__bf16*)(ws + carve((size_t)BB * LL * HH * 2));        // leaves
  __bf16* bufB  = (__bf16*)(ws + carve((size_t)BB * (LL / 2) * HH * 2));  // ping-pong

  // 1) weights -> bf16
  cvt_bf16_kernel<<<(K3H * K3H + 255) / 256, 256, 0, stream>>>(Wr, Wr_bf, K3H * K3H);
  cvt_bf16_kernel<<<(HH * K3H + 255) / 256, 256, 0, stream>>>(Wh, Wh_bf, HH * K3H);
  cvt_bf16_kernel<<<(K4H * K4H + 255) / 256, 256, 0, stream>>>(Wz, Wz_bf, K4H * K4H);

  // 2) leaf projection -> bufA
  leaf_kernel<<<BB * LL, HH, 0, stream>>>(data, Wu, bu, bufA);

  // 3) tree levels l = 8..0 (ping-pong bufA/bufB); rows = 256*2^l, /32 per block
  __bf16* cur = bufA;  // holds 2^(l+1) nodes entering level l
  __bf16* nxt = bufB;
  for (int l = 8; l >= 0; --l) {
    int n = 1 << l;
    level_kernel<<<8 * n, 512, 0, stream>>>(data, Wu, bu, Wr_bf, br, Wh_bf,
                                            bh, Wz_bf, bz, cur, nxt, n, l);
    __bf16* tmp = cur; cur = nxt; nxt = tmp;
  }

  // 4) head -> d_out (f32, B elements); `cur` holds level-0 output (1 node)
  head_kernel<<<BB, 32, 0, stream>>>(cur, Wp, bp, (float*)d_out);
}
